// GatedConv_44006234915419
// MI455X (gfx1250) — matrix operand order repaired and checked by reading, not compiled, not run on hardware
//
#include <hip/hip_runtime.h>
#include <hip/hip_bf16.h>
#include <stddef.h>
#include <stdint.h>

typedef __attribute__((ext_vector_type(16))) _Float16 v16h;
typedef __attribute__((ext_vector_type(8)))  _Float16 v8h;
typedef __attribute__((ext_vector_type(8)))  float    v8f;

#define CIN   256
#define COUT  256
#define HD    56
#define WD    56
#define HW    3136      // 56*56
#define KTOT  2304      // 9*256
#define BATCH 32
#define NPOS  100352    // 32*3136
#define YELEMS 25690112 // 32*256*3136

// ---- CDNA5 async global->LDS copy (ASYNCcnt-tracked), ISA §15.18.3 op 98 ----
__device__ __forceinline__ void async_copy_b128(unsigned lds_addr, const void* gaddr) {
    asm volatile("global_load_async_to_lds_b128 %0, %1, off"
                 :: "v"(lds_addr), "v"((unsigned long long)(uintptr_t)gaddr)
                 : "memory");
}
__device__ __forceinline__ void wait_async0() {
    asm volatile("s_wait_asynccnt 0x0" ::: "memory");
}

// ---------------- weight repack: conv_w[co][ci][kh][kw] -> Wh[co][tap*256+ci] (f16) ----------------
__global__ __launch_bounds__(256)
void wcast_kernel(const float* __restrict__ cw, _Float16* __restrict__ wh) {
    int idx = blockIdx.x * 256 + threadIdx.x;          // < 256*2304
    int co = idx / KTOT;
    int k  = idx - co * KTOT;
    int tap = k >> 8;                                  // 0..8
    int ci  = k & 255;
    wh[idx] = (_Float16)cw[(size_t)(co * CIN + ci) * 9 + tap];
}

// ---------------- global average pool: pooled[b*256+ci] ----------------
__global__ __launch_bounds__(256)
void pool_kernel(const float* __restrict__ x, float* __restrict__ pooled) {
    __shared__ float red[256];
    int bc = blockIdx.x;                               // b*256+ci
    const float* p = x + (size_t)bc * HW;
    float s = 0.0f;
    for (int i = threadIdx.x; i < HW; i += 256) s += p[i];
    red[threadIdx.x] = s;
    __syncthreads();
    for (int o = 128; o > 0; o >>= 1) {
        if (threadIdx.x < (unsigned)o) red[threadIdx.x] += red[threadIdx.x + o];
        __syncthreads();
    }
    if (threadIdx.x == 0) pooled[bc] = red[0] * (1.0f / (float)HW);
}

// ---------------- gates: relu(pooled @ gate_w^T + b), keep top-128 per sample ----------------
__global__ __launch_bounds__(256)
void gate_kernel(const float* __restrict__ pooled, const float* __restrict__ gw,
                 const float* __restrict__ gb, float* __restrict__ gates) {
    __shared__ float pr[256];
    __shared__ float gv[256];
    int b  = blockIdx.x;
    int co = threadIdx.x;
    pr[co] = pooled[b * 256 + co];
    __syncthreads();
    const float* wrow = gw + (size_t)co * 256;
    float d = 0.0f;
    #pragma unroll 8
    for (int i = 0; i < 256; ++i) d += pr[i] * wrow[i];
    float g = fmaxf(d + gb[co], 0.0f);
    gv[co] = g;
    __syncthreads();
    // zeroed set = first 128 in (value asc, index asc) order -> keep if rank >= 128
    int rank = 0;
    for (int c = 0; c < 256; ++c)
        rank += (gv[c] < g) || ((gv[c] == g) && (c < co));
    gates[b * 256 + co] = (rank >= 128) ? g : 0.0f;
}

// ---------------- implicit-GEMM 3x3 conv via v_wmma_f32_16x16x32_f16 ----------------
// Grid: (NPOS/64, COUT/128). WG = 256 thr = 8 waves.
// Wave (mg,ng): 32x32 output block = 2x2 WMMA subtiles.
// A panel: async global->LDS (double-buffered, ASYNCcnt). B panel: reg double-buffer.
// One s_barrier per K-step.
__global__ __launch_bounds__(256)
void conv_wmma_kernel(const float* __restrict__ x,
                      const _Float16* __restrict__ Wh,
                      float* __restrict__ y) {
    __shared__ __align__(64) _Float16 sA[2][128][32];   // [buf][co_local][k_local]  16 KB
    __shared__ __align__(64) _Float16 sB[2][64][32];    // [buf][pos_local][k_local]  8 KB

    const int t    = threadIdx.x;
    const int lane = t & 31;
    const int wave = t >> 5;
    const int mg   = wave & 3;        // M group (32 couts)
    const int ng   = wave >> 2;       // N group (32 positions)
    const int coBase = blockIdx.y * 128;
    const int pBase  = blockIdx.x * 64;

    // ---- staging roles ----
    // A: 128 rows x 32 k; thread -> one 32B chunk (2 async b128 copies)
    const int aRow = t >> 1;            // 0..127
    const int aCol = (t & 1) * 16;      // 0 or 16
    // B: 64 pos x 32 k; thread -> fixed pos, 8 consecutive k (lane-consecutive positions)
    const int bPos = t & 63;
    const int bKg  = t >> 6;            // 0..3 -> k = bKg*8 + j

    const int pS   = pBase + bPos;      // tile never crosses images (3136 % 64 == 0)
    const int bImg = pS / HW;
    const int hwS  = pS - bImg * HW;
    const int hS   = hwS / WD;
    const int wS   = hwS - hS * WD;

    const _Float16* wPtr  = Wh + (size_t)(coBase + aRow) * KTOT + aCol;
    const float*    xBase = x + (size_t)bImg * CIN * HW;

    const unsigned ldsA0 = (unsigned)(uintptr_t)&sA[0][aRow][aCol];
    const unsigned ldsA1 = (unsigned)(uintptr_t)&sA[1][aRow][aCol];

    v8f acc00 = {}, acc01 = {}, acc10 = {}, acc11 = {};

    // ---- prologue: A(k=0) async into sA[0]; B(k=0) gathered into registers ----
    async_copy_b128(ldsA0,      wPtr);
    async_copy_b128(ldsA0 + 16, wPtr + 8);
    float bReg[8];
    {
        const int hh = hS - 1, ww = wS - 1;      // tap 0: dh=dw=-1
        const bool ok = (hh >= 0) && (ww >= 0);
        const float* xp = xBase + (size_t)(bKg * 8) * HW + hh * WD + ww;
        #pragma unroll
        for (int j = 0; j < 8; ++j) bReg[j] = ok ? xp[(size_t)j * HW] : 0.0f;
    }

    int buf = 0;
    for (int kb = 0; kb < KTOT; kb += 32) {
        // ---- commit B registers into sB[buf] ----
        {
            _Float16* dst = &sB[buf][bPos][bKg * 8];
            #pragma unroll
            for (int j = 0; j < 8; ++j) dst[j] = (_Float16)bReg[j];
        }
        // A(kb) async copies must have landed in LDS before the barrier
        wait_async0();
        __syncthreads();

        // ---- after the barrier it is safe to overwrite the other buffer:
        //      issue async A(kb+32) and gather B(kb+32); overlaps with WMMAs ----
        const int kn = kb + 32;
        if (kn < KTOT) {
            const unsigned ldsN = buf ? ldsA0 : ldsA1;
            async_copy_b128(ldsN,      wPtr + kn);
            async_copy_b128(ldsN + 16, wPtr + kn + 8);

            const int tap = kn >> 8;
            const int ciB = (kn & 255) + bKg * 8;
            const int dh  = tap / 3 - 1;
            const int dw  = tap - (tap / 3) * 3 - 1;
            const int hh  = hS + dh, ww = wS + dw;
            const bool ok = (hh >= 0) && (hh < HD) && (ww >= 0) && (ww < WD);
            const float* xp = xBase + (size_t)ciB * HW + hh * WD + ww;
            #pragma unroll
            for (int j = 0; j < 8; ++j) bReg[j] = ok ? xp[(size_t)j * HW] : 0.0f;
        }

        // ---- wave compute: 2x2 subtiles, fragments reused 2x each ----
        {
            const int khalf = lane >> 4;
            const int l15   = lane & 15;

            v16h a0, a1;
            {
                const int mr0 = mg * 32 + l15;
                const int mr1 = mr0 + 16;
                v8h lo0 = *(const v8h*)&sA[buf][mr0][khalf * 16];
                v8h hi0 = *(const v8h*)&sA[buf][mr0][khalf * 16 + 8];
                v8h lo1 = *(const v8h*)&sA[buf][mr1][khalf * 16];
                v8h hi1 = *(const v8h*)&sA[buf][mr1][khalf * 16 + 8];
                #pragma unroll
                for (int i = 0; i < 8; ++i) {
                    a0[i] = lo0[i]; a0[8 + i] = hi0[i];
                    a1[i] = lo1[i]; a1[8 + i] = hi1[i];
                }
            }
            const int nr0 = ng * 32 + l15;
            v16h b0 = *(const v16h*)&sB[buf][nr0][khalf * 16];
            v16h b1 = *(const v16h*)&sB[buf][nr0 + 16][khalf * 16];

            acc00 = __builtin_amdgcn_wmma_f32_16x16x32_f16(false, a0, false, b0, (short)0, acc00, false, false);
            acc01 = __builtin_amdgcn_wmma_f32_16x16x32_f16(false, a0, false, b1, (short)0, acc01, false, false);
            acc10 = __builtin_amdgcn_wmma_f32_16x16x32_f16(false, a1, false, b0, (short)0, acc10, false, false);
            acc11 = __builtin_amdgcn_wmma_f32_16x16x32_f16(false, a1, false, b1, (short)0, acc11, false, false);
        }
        buf ^= 1;
    }

    // ---- write C: VGPR r -> M = r + 8*(lane>=16), N = lane&15 ----
    {
        const int n     = lane & 15;
        const int khalf = lane >> 4;
        const v8f* accs[4] = { &acc00, &acc01, &acc10, &acc11 };
        #pragma unroll
        for (int si = 0; si < 2; ++si) {
            #pragma unroll
            for (int sj = 0; sj < 2; ++sj) {
                const v8f& a = *accs[si * 2 + sj];
                const int p  = pBase + ng * 32 + sj * 16 + n;
                const int bo = p / HW;
                const int hw = p - bo * HW;
                #pragma unroll
                for (int r = 0; r < 8; ++r) {
                    int co = coBase + mg * 32 + si * 16 + r + 8 * khalf;
                    y[(size_t)(bo * COUT + co) * HW + hw] = a[r];
                }
            }
        }
    }
}

// ---------------- per-channel batch stats: partial pass (grid 256 x 8) ----------------
__global__ __launch_bounds__(256)
void bnstat_part_kernel(const float* __restrict__ y, float* __restrict__ psum,
                        float* __restrict__ qsum) {
    __shared__ float rs[256];
    __shared__ float rq[256];
    int c = blockIdx.x;
    int chunk = blockIdx.y;              // 4 batch images per chunk
    float s = 0.0f, q = 0.0f;
    for (int bb = 0; bb < 4; ++bb) {
        int b = chunk * 4 + bb;
        const float* p = y + (size_t)(b * COUT + c) * HW;
        for (int i = threadIdx.x; i < HW; i += 256) {
            float v = p[i];
            s += v; q += v * v;
        }
    }
    rs[threadIdx.x] = s; rq[threadIdx.x] = q;
    __syncthreads();
    for (int o = 128; o > 0; o >>= 1) {
        if (threadIdx.x < (unsigned)o) {
            rs[threadIdx.x] += rs[threadIdx.x + o];
            rq[threadIdx.x] += rq[threadIdx.x + o];
        }
        __syncthreads();
    }
    if (threadIdx.x == 0) {
        psum[c * 8 + chunk] = rs[0];
        qsum[c * 8 + chunk] = rq[0];
    }
}

// ---------------- finalize mean / biased var (1 block) ----------------
__global__ __launch_bounds__(256)
void bnfin_kernel(const float* __restrict__ psum, const float* __restrict__ qsum,
                  float* __restrict__ mean, float* __restrict__ var) {
    int c = threadIdx.x;
    float s = 0.0f, q = 0.0f;
    #pragma unroll
    for (int i = 0; i < 8; ++i) { s += psum[c * 8 + i]; q += qsum[c * 8 + i]; }
    float m = s * (1.0f / (float)NPOS);
    mean[c] = m;
    var[c]  = q * (1.0f / (float)NPOS) - m * m;
}

// ---------------- fused BN * gate -> relu (float4) ----------------
__global__ __launch_bounds__(256)
void final_kernel(const float* __restrict__ y, const float* __restrict__ mean,
                  const float* __restrict__ var, const float* __restrict__ gamma,
                  const float* __restrict__ beta, const float* __restrict__ gates,
                  float* __restrict__ out) {
    int v4 = blockIdx.x * 256 + threadIdx.x;
    if (v4 >= YELEMS / 4) return;
    int e = v4 * 4;                     // 3136 % 4 == 0 -> stays in one (b,c) plane
    int plane = e / HW;                 // b*256 + c
    int c = plane & 255;
    float sc = gamma[c] * rsqrtf(var[c] + 1e-5f);
    float sh = beta[c] - mean[c] * sc;
    float g  = gates[plane];
    float4 v = *(const float4*)(y + e);
    float4 o;
    o.x = fmaxf((v.x * sc + sh) * g, 0.0f);
    o.y = fmaxf((v.y * sc + sh) * g, 0.0f);
    o.z = fmaxf((v.z * sc + sh) * g, 0.0f);
    o.w = fmaxf((v.w * sc + sh) * g, 0.0f);
    *(float4*)(out + e) = o;
}

extern "C" void kernel_launch(void* const* d_in, const int* in_sizes, int n_in,
                              void* d_out, int out_size, void* d_ws, size_t ws_size,
                              hipStream_t stream) {
    const float* x      = (const float*)d_in[0];
    const float* conv_w = (const float*)d_in[1];
    const float* gate_w = (const float*)d_in[2];
    const float* gate_b = (const float*)d_in[3];
    const float* gamma  = (const float*)d_in[4];
    const float* beta   = (const float*)d_in[5];
    float* out = (float*)d_out;

    // workspace layout
    char* ws = (char*)d_ws;
    float*     yws    = (float*)ws;                                  // 25690112 f32
    _Float16*  Wh     = (_Float16*)(ws + (size_t)YELEMS * 4);        // 589824 f16
    float*     pooled = (float*)((char*)Wh + (size_t)COUT * KTOT * 2);
    float*     gates  = pooled + BATCH * CIN;
    float*     meanp  = gates + BATCH * COUT;
    float*     varp   = meanp + COUT;
    float*     psum   = varp + COUT;        // 256*8
    float*     qsum   = psum + COUT * 8;    // 256*8

    wcast_kernel<<<(COUT * KTOT) / 256, 256, 0, stream>>>(conv_w, Wh);
    pool_kernel<<<BATCH * CIN, 256, 0, stream>>>(x, pooled);
    gate_kernel<<<BATCH, 256, 0, stream>>>(pooled, gate_w, gate_b, gates);
    conv_wmma_kernel<<<dim3(NPOS / 64, COUT / 128), 256, 0, stream>>>(x, Wh, yws);
    bnstat_part_kernel<<<dim3(COUT, 8), 256, 0, stream>>>(yws, psum, qsum);
    bnfin_kernel<<<1, 256, 0, stream>>>(psum, qsum, meanp, varp);
    final_kernel<<<(YELEMS / 4 + 255) / 256, 256, 0, stream>>>(yws, meanp, varp,
                                                               gamma, beta, gates, out);
}
